// KPRN_80590766342901
// MI455X (gfx1250) — matrix operand order repaired and checked by reading, not compiled, not run on hardware
//
#include <hip/hip_runtime.h>
#include <hip/hip_bf16.h>

typedef __attribute__((ext_vector_type(16))) _Float16 v16h;
typedef __attribute__((ext_vector_type(8)))  _Float16 v8h;
typedef __attribute__((ext_vector_type(8)))  float    v8f;

#define BB 64
#define CC 5
#define PP 5
#define LL 5
#define DD 128
#define TT 768
#define NNODE 8000   // B*C*P*L
#define NBC 320      // B*C
#define GG 512       // 4*D gates
#define K0 384       // 3*D lstm0 input

union U16h { v16h v; v8h h[2]; };

// ---- WMMA fragment loaders (v_wmma_f32_16x16x32_f16 layouts) ----
// A 16x32 f16, row-major source, lda in halves. lane l: m=l&15, k-half h=l>>4.
// elem e: k = (e&7) + ((e>>3)<<4) + h*8  -> two contiguous 16B chunks.
__device__ __forceinline__ v16h frag_a_global(const _Float16* A, int lda) {
  int lane = threadIdx.x & 31;
  const _Float16* p = A + (size_t)(lane & 15) * lda + (lane >> 4) * 8;
  U16h u; u.h[0] = *(const v8h*)p; u.h[1] = *(const v8h*)(p + 16);
  return u.v;
}
// B 32x16: B[k][n] = W[n][k] with W row-major [N][K] (torch [out][in]).
// lane l: n=l&15, elem e: k = (e&7) + ((e>>3)<<4) + 8*(l>>4) -> two contiguous 16B chunks.
__device__ __forceinline__ v16h frag_b_global(const _Float16* W, int ldb) {
  int lane = threadIdx.x & 31;
  const _Float16* p = W + (size_t)(lane & 15) * ldb + (lane >> 4) * 8;
  U16h u; u.h[0] = *(const v8h*)p; u.h[1] = *(const v8h*)(p + 16);
  return u.v;
}
__device__ __forceinline__ v8f wmma_f16(v16h a, v16h b, v8f c) {
  return __builtin_amdgcn_wmma_f32_16x16x32_f16(false, a, false, b, (short)0, c, false, false);
}
__device__ __forceinline__ v8h cvt8(const float* p) {
  float4 a = ((const float4*)p)[0], b = ((const float4*)p)[1];
  v8h r;
  r[0]=(_Float16)a.x; r[1]=(_Float16)a.y; r[2]=(_Float16)a.z; r[3]=(_Float16)a.w;
  r[4]=(_Float16)b.x; r[5]=(_Float16)b.y; r[6]=(_Float16)b.z; r[7]=(_Float16)b.w;
  return r;
}
__device__ __forceinline__ float sigmoidf_(float x) { return 1.0f / (1.0f + expf(-x)); }

// ---- utility kernels ----
__global__ void k_cast(_Float16* dst, const float* src, int n) {
  int i = blockIdx.x * 256 + threadIdx.x;
  if (i < n) dst[i] = (_Float16)src[i];
}
__global__ void k_bias(float* dst, const float* a, const float* b, int n) {
  int i = blockIdx.x * 256 + threadIdx.x;
  if (i < n) dst[i] = a[i] + b[i];
}
__global__ void k_zero(unsigned int* p, int n) {
  int i = blockIdx.x * 256 + threadIdx.x;
  if (i < n) p[i] = 0u;
}

// ---- gather indices, build X[p][n][l][384] (type/rel segments + user/entity node emb) ----
__global__ void k_build(const int* user_index, const int* paths, const int* types, const int* rels,
                        const float* user_e, const float* ent_e, const float* type_e, const float* rel_e,
                        _Float16* X, int* nidx, int* ntype) {
  int tid = blockIdx.x * 256 + threadIdx.x;   // NNODE * 128 threads
  int node = tid >> 7, d = tid & 127;
  if (node >= NNODE) return;
  int l = node % 5, p = (node / 5) % 5, c = (node / 25) % 5, b = node / 125;
  int u = user_index[b];
  int off = ((u * 5 + c) * 5 + p) * 5 + l;
  int idx = paths[off], ty = types[off], rl = rels[off];
  int rx = (p * NBC + b * 5 + c) * 5 + l;
  _Float16* xr = X + (size_t)rx * K0;
  xr[128 + d] = (_Float16)type_e[ty * DD + d];
  xr[256 + d] = (_Float16)rel_e[rl * DD + d];
  if (ty == 0)      xr[d] = (_Float16)user_e[(size_t)idx * DD + d];
  else if (ty == 2) xr[d] = (_Float16)ent_e[(size_t)idx * DD + d];
  if (d == 0) { nidx[node] = idx; ntype[node] = ty; }
}

// ---- news MLP layer1: hid = ELU(gather(title) @ nc1^T + b), [8000,768]x[768->128] ----
__global__ void k_news1(const int* nidx, const float* title, const _Float16* nc1h,
                        const float* nc1b, _Float16* hid) {
  int mt = blockIdx.x;                 // 500 M-tiles
  int nt = threadIdx.x >> 5;           // 8 waves = 8 N-tiles (128 cols)
  int lane = threadIdx.x & 31;
  int hh = lane >> 4, nn = lane & 15;
  const float* rowp = title + (size_t)nidx[mt * 16 + nn] * TT;  // gathered A row (m = lane&15)
  const _Float16* Bb = nc1h + (size_t)(nt * 16) * TT;
  v8f acc = {};
  for (int kk = 0; kk < TT / 32; ++kk) {
    U16h ua;
    ua.h[0] = cvt8(rowp + kk * 32 + hh * 8);
    ua.h[1] = cvt8(rowp + kk * 32 + 16 + hh * 8);
    acc = wmma_f16(ua.v, frag_b_global(Bb + kk * 32, TT), acc);
  }
#pragma unroll
  for (int j = 0; j < 8; ++j) {
    int m = j + 8 * hh, gc = nt * 16 + nn;
    float v = acc[j] + nc1b[gc];
    v = v > 0.f ? v : (expf(v) - 1.0f);            // ELU
    hid[(size_t)(mt * 16 + m) * DD + gc] = (_Float16)v;
  }
}

// ---- news MLP layer2: X[node][0:128] = tanh(hid @ nc2^T + b) where type==1 ----
__global__ void k_news2(const int* ntype, const _Float16* hid, const _Float16* nc2h,
                        const float* nc2b, _Float16* X) {
  int mt = blockIdx.x;
  int nt = threadIdx.x >> 5;
  int lane = threadIdx.x & 31;
  int hh = lane >> 4, nn = lane & 15;
  const _Float16* Ab = hid + (size_t)(mt * 16) * DD;
  const _Float16* Bb = nc2h + (size_t)(nt * 16) * DD;
  v8f acc = {};
  for (int kk = 0; kk < DD / 32; ++kk)
    acc = wmma_f16(frag_a_global(Ab + kk * 32, DD), frag_b_global(Bb + kk * 32, DD), acc);
#pragma unroll
  for (int j = 0; j < 8; ++j) {
    int m = j + 8 * hh, gc = nt * 16 + nn;
    int node = mt * 16 + m;
    if (ntype[node] == 1) {
      int l = node % 5, p = (node / 5) % 5, c = (node / 25) % 5, b = node / 125;
      int rx = (p * NBC + b * 5 + c) * 5 + l;
      X[(size_t)rx * K0 + gc] = (_Float16)tanhf(acc[j] + nc2b[gc]);
    }
  }
}

// ---- pre0[p][n*5+l][512] = X[p] @ wih0[p]^T + (bih0+bhh0) ----
__global__ void k_pre0(const _Float16* X, const _Float16* wih0h, const float* b0, float* pre0) {
  int mt = blockIdx.x;                            // 100 M-tiles over 1600 rows
  int p = blockIdx.y;                             // 5 paths
  int nt = blockIdx.z * 8 + (threadIdx.x >> 5);   // 32 N-tiles over 512 gates
  int lane = threadIdx.x & 31;
  int hh = lane >> 4, nn = lane & 15;
  const _Float16* Ab = X + (size_t)p * 1600 * K0 + (size_t)(mt * 16) * K0;
  const _Float16* Bb = wih0h + (size_t)p * GG * K0 + (size_t)(nt * 16) * K0;
  v8f acc = {};
  for (int kk = 0; kk < K0 / 32; ++kk)
    acc = wmma_f16(frag_a_global(Ab + kk * 32, K0), frag_b_global(Bb + kk * 32, K0), acc);
#pragma unroll
  for (int j = 0; j < 8; ++j) {
    int r = mt * 16 + j + 8 * hh, gc = nt * 16 + nn;
    pre0[((size_t)p * 1600 + r) * GG + gc] = acc[j] + b0[p * GG + gc];
  }
}

// ---- one LSTM timestep, both layers fused; block = (mt 16-row band, path p) ----
__global__ void k_step(int t, const float* pre0, const _Float16* whh0h, const _Float16* wih1h,
                       const _Float16* whh1h, const float* b1,
                       _Float16* h0w, _Float16* h1w, float* c0w, float* c1w, float* sel) {
  int p = blockIdx.y;
  if (p < t) return;                              // path p only needs steps 0..p
  int mt = blockIdx.x;                            // 20 M-tiles over 320 rows
  int wave = threadIdx.x >> 5, lane = threadIdx.x & 31;
  int hh = lane >> 4, nn = lane & 15;
  __shared__ float gbuf[16][GG];                  // 32 KB gate staging
  __shared__ _Float16 h0t[16][DD];                // 4 KB layer0 hidden tile

  // phase 0: layer0 gates = h0_prev @ whh0^T + pre0[:, t]
  const _Float16* Ah = h0w + ((size_t)p * NBC + mt * 16) * DD;
  for (int i = 0; i < 4; ++i) {
    int nt = wave * 4 + i;
    const _Float16* Bb = whh0h + (size_t)p * GG * DD + (size_t)(nt * 16) * DD;
    v8f acc = {};
    for (int kk = 0; kk < 4; ++kk)
      acc = wmma_f16(frag_a_global(Ah + kk * 32, DD), frag_b_global(Bb + kk * 32, DD), acc);
#pragma unroll
    for (int j = 0; j < 8; ++j) {
      int m = j + 8 * hh, gc = nt * 16 + nn;
      int r = mt * 16 + m;
      gbuf[m][gc] = acc[j] + pre0[((size_t)p * 1600 + r * 5 + t) * GG + gc];
    }
  }
  __syncthreads();
  // phase 1: layer0 cell update
  for (int u = threadIdx.x; u < 16 * DD; u += 256) {
    int m = u >> 7, hu = u & 127;
    int r = mt * 16 + m;
    size_t ci = ((size_t)p * NBC + r) * DD + hu;
    float ig = gbuf[m][hu], fg = gbuf[m][DD + hu], gg = gbuf[m][2 * DD + hu], og = gbuf[m][3 * DD + hu];
    float c = sigmoidf_(fg) * c0w[ci] + sigmoidf_(ig) * tanhf(gg);
    float h = sigmoidf_(og) * tanhf(c);
    c0w[ci] = c;
    _Float16 hf = (_Float16)h;
    h0w[ci] = hf;
    h0t[m][hu] = hf;
  }
  __syncthreads();
  // phase 2: layer1 gates = h0_new(LDS) @ wih1^T + h1_prev @ whh1^T + b1
  const _Float16* A2 = h1w + ((size_t)p * NBC + mt * 16) * DD;
  for (int i = 0; i < 4; ++i) {
    int nt = wave * 4 + i;
    const _Float16* B1 = wih1h + (size_t)p * GG * DD + (size_t)(nt * 16) * DD;
    const _Float16* B2 = whh1h + (size_t)p * GG * DD + (size_t)(nt * 16) * DD;
    v8f acc = {};
    for (int kk = 0; kk < 4; ++kk) {
      const _Float16* pa = &h0t[nn][kk * 32 + hh * 8];   // A fragment straight from LDS
      U16h ua; ua.h[0] = *(const v8h*)pa; ua.h[1] = *(const v8h*)(pa + 16);
      acc = wmma_f16(ua.v, frag_b_global(B1 + kk * 32, DD), acc);
      acc = wmma_f16(frag_a_global(A2 + kk * 32, DD), frag_b_global(B2 + kk * 32, DD), acc);
    }
#pragma unroll
    for (int j = 0; j < 8; ++j) {
      int m = j + 8 * hh, gc = nt * 16 + nn;
      gbuf[m][gc] = acc[j] + b1[p * GG + gc];
    }
  }
  __syncthreads();
  // phase 3: layer1 cell update (+ capture output at t == p)
  for (int u = threadIdx.x; u < 16 * DD; u += 256) {
    int m = u >> 7, hu = u & 127;
    int r = mt * 16 + m;
    size_t ci = ((size_t)p * NBC + r) * DD + hu;
    float ig = gbuf[m][hu], fg = gbuf[m][DD + hu], gg = gbuf[m][2 * DD + hu], og = gbuf[m][3 * DD + hu];
    float c = sigmoidf_(fg) * c1w[ci] + sigmoidf_(ig) * tanhf(gg);
    float h = sigmoidf_(og) * tanhf(c);
    c1w[ci] = c;
    h1w[ci] = (_Float16)h;
    if (t == p) sel[((size_t)r * PP + p) * DD + hu] = h;
  }
}

// ---- path scorer + softmax over C ----
__global__ void k_score(const float* sel, const float* ps1w, const float* ps1b,
                        const float* ps2w, const float* ps2b, float* out) {
  int b = blockIdx.x;
  __shared__ float sc[CC];
  if (threadIdx.x < CC) sc[threadIdx.x] = 0.f;
  __syncthreads();
  int wave = threadIdx.x >> 5, lane = threadIdx.x & 31;
  for (int q = wave; q < CC * PP; q += 8) {
    int c = q / PP, pp = q % PP;
    const float* x = sel + ((size_t)(b * CC + c) * PP + pp) * DD;
    float part = 0.f;
    if (lane < 16) {
      float s = 0.f;
      for (int d = 0; d < DD; ++d) s += ps1w[lane * DD + d] * x[d];
      s += ps1b[lane];
      s = s > 0.f ? s : 0.f;
      part = s * ps2w[lane];
    }
    for (int o = 8; o >= 1; o >>= 1) part += __shfl_xor(part, o, 16);
    if (lane == 0) atomicAdd(&sc[c], part + ps2b[0]);
  }
  __syncthreads();
  if (threadIdx.x == 0) {
    float mx = sc[0];
    for (int c = 1; c < CC; ++c) mx = fmaxf(mx, sc[c]);
    float e[CC], sum = 0.f;
    for (int c = 0; c < CC; ++c) { e[c] = expf(sc[c] - mx); sum += e[c]; }
    for (int c = 0; c < CC; ++c) out[b * CC + c] = e[c] / sum;
  }
}

extern "C" void kernel_launch(void* const* d_in, const int* in_sizes, int n_in,
                              void* d_out, int out_size, void* d_ws, size_t ws_size,
                              hipStream_t stream) {
  const int*   user_index = (const int*)d_in[1];
  const int*   paths = (const int*)d_in[2];
  const int*   types = (const int*)d_in[3];
  const int*   rels  = (const int*)d_in[4];
  const float* title = (const float*)d_in[5];
  const float* ent_e = (const float*)d_in[6];
  const float* rel_e = (const float*)d_in[7];
  const float* user_e = (const float*)d_in[8];
  const float* type_e = (const float*)d_in[9];
  const float* nc1w = (const float*)d_in[10];
  const float* nc1b = (const float*)d_in[11];
  const float* nc2w = (const float*)d_in[12];
  const float* nc2b = (const float*)d_in[13];
  const float* ps1w = (const float*)d_in[14];
  const float* ps1b = (const float*)d_in[15];
  const float* ps2w = (const float*)d_in[16];
  const float* ps2b = (const float*)d_in[17];
  const float* wih0 = (const float*)d_in[18];
  const float* whh0 = (const float*)d_in[19];
  const float* bih0 = (const float*)d_in[20];
  const float* bhh0 = (const float*)d_in[21];
  const float* wih1 = (const float*)d_in[22];
  const float* whh1 = (const float*)d_in[23];
  const float* bih1 = (const float*)d_in[24];
  const float* bhh1 = (const float*)d_in[25];

  char* w = (char*)d_ws; size_t off = 0;
  auto alloc = [&](size_t bytes) { void* r = w + off; off += (bytes + 255) & ~(size_t)255; return r; };
  _Float16* nc1h  = (_Float16*)alloc((size_t)DD * TT * 2);
  _Float16* nc2h  = (_Float16*)alloc((size_t)DD * DD * 2);
  _Float16* wih0h = (_Float16*)alloc((size_t)PP * GG * K0 * 2);
  _Float16* whh0h = (_Float16*)alloc((size_t)PP * GG * DD * 2);
  _Float16* wih1h = (_Float16*)alloc((size_t)PP * GG * DD * 2);
  _Float16* whh1h = (_Float16*)alloc((size_t)PP * GG * DD * 2);
  float*    b0    = (float*)alloc((size_t)PP * GG * 4);
  float*    b1    = (float*)alloc((size_t)PP * GG * 4);
  _Float16* X     = (_Float16*)alloc((size_t)NNODE * K0 * 2);
  _Float16* hid   = (_Float16*)alloc((size_t)NNODE * DD * 2);
  float*    pre0  = (float*)alloc((size_t)PP * 1600 * GG * 4);
  _Float16* h0w   = (_Float16*)alloc((size_t)PP * NBC * DD * 2);   // zeroed region start
  _Float16* h1w   = (_Float16*)alloc((size_t)PP * NBC * DD * 2);
  float*    c0w   = (float*)alloc((size_t)PP * NBC * DD * 4);
  float*    c1w   = (float*)alloc((size_t)PP * NBC * DD * 4);      // zeroed region end
  float*    sel   = (float*)alloc((size_t)NBC * PP * DD * 4);
  int*      nidx  = (int*)alloc((size_t)NNODE * 4);
  int*      ntype = (int*)alloc((size_t)NNODE * 4);

  auto cast = [&](_Float16* d, const float* s, int n) {
    k_cast<<<(n + 255) / 256, 256, 0, stream>>>(d, s, n);
  };
  cast(nc1h, nc1w, DD * TT);
  cast(nc2h, nc2w, DD * DD);
  cast(wih0h, wih0, PP * GG * K0);
  cast(whh0h, whh0, PP * GG * DD);
  cast(wih1h, wih1, PP * GG * DD);
  cast(whh1h, whh1, PP * GG * DD);
  k_bias<<<(PP * GG + 255) / 256, 256, 0, stream>>>(b0, bih0, bhh0, PP * GG);
  k_bias<<<(PP * GG + 255) / 256, 256, 0, stream>>>(b1, bih1, bhh1, PP * GG);

  k_build<<<(NNODE * 128) / 256, 256, 0, stream>>>(user_index, paths, types, rels,
                                                   user_e, ent_e, type_e, rel_e, X, nidx, ntype);
  k_news1<<<NNODE / 16, 256, 0, stream>>>(nidx, title, nc1h, nc1b, hid);
  k_news2<<<NNODE / 16, 256, 0, stream>>>(ntype, hid, nc2h, nc2b, X);
  k_pre0<<<dim3(100, PP, 4), 256, 0, stream>>>(X, wih0h, b0, pre0);

  int zeroInts = (int)((size_t)PP * NBC * DD * (2 + 2 + 4 + 4) / 4);  // h0,h1 f16 + c0,c1 f32
  k_zero<<<(zeroInts + 255) / 256, 256, 0, stream>>>((unsigned int*)h0w, zeroInts);

  for (int t = 0; t < LL; ++t)
    k_step<<<dim3(NBC / 16, PP), 256, 0, stream>>>(t, pre0, whh0h, wih1h, whh1h, b1,
                                                   h0w, h1w, c0w, c1w, sel);

  k_score<<<BB, 256, 0, stream>>>(sel, ps1w, ps1b, ps2w, ps2b, (float*)d_out);
}
